// _ThreeWayAttention_49134425866852
// MI455X (gfx1250) — compile-verified, matrix operand order checked by reading, not run on hardware
//
#include <hip/hip_runtime.h>
#include <stdint.h>

#define NN 8192
#define DD 256

typedef __attribute__((ext_vector_type(16))) __bf16 v16bf;
typedef __attribute__((ext_vector_type(8)))  float  v8f;
typedef __attribute__((ext_vector_type(4)))  int    v4i;

// ---------- CDNA5 async-to-LDS staging (guarded; falls back to ds copies) ----
#if defined(__has_builtin)
#if __has_builtin(__builtin_amdgcn_global_load_async_to_lds_b128) && \
    __has_builtin(__builtin_amdgcn_s_wait_asynccnt)
#define HAVE_ASYNC_LDS 1
#endif
#endif

static __device__ __forceinline__ void cp16(void* l, const void* g) {
#ifdef HAVE_ASYNC_LDS
  __builtin_amdgcn_global_load_async_to_lds_b128(
      (__attribute__((address_space(1))) v4i*)g,
      (__attribute__((address_space(3))) v4i*)l, 0, 0);
#else
  *(uint4*)l = *(const uint4*)g;
#endif
}
static __device__ __forceinline__ void cp_fence() {
#ifdef HAVE_ASYNC_LDS
  __builtin_amdgcn_s_wait_asynccnt(0);
#endif
}

// ---------- helpers ----------
static __device__ __forceinline__ unsigned short f2bfu(float f) {
  unsigned u = __builtin_bit_cast(unsigned, f);
  return (unsigned short)((u + 0x7FFFu + ((u >> 16) & 1u)) >> 16);  // RNE
}
static __device__ __forceinline__ __bf16 ubf(unsigned short s) {
  return __builtin_bit_cast(__bf16, s);
}
// 16-bit A/B operand element->K mapping (wave32, ISA 7.12.2)
static __device__ __forceinline__ int kmap(int e, int hf) {
  int j = e >> 1, p = e & 1;
  return ((j >> 2) << 4) + (hf << 3) + ((j & 3) << 1) + p;
}
static __device__ __forceinline__ float rowmax16(float v) {
  v = fmaxf(v, __shfl_xor(v, 1, 32));
  v = fmaxf(v, __shfl_xor(v, 2, 32));
  v = fmaxf(v, __shfl_xor(v, 4, 32));
  v = fmaxf(v, __shfl_xor(v, 8, 32));
  return v;
}
static __device__ __forceinline__ float rowsum16(float v) {
  v += __shfl_xor(v, 1, 32);
  v += __shfl_xor(v, 2, 32);
  v += __shfl_xor(v, 4, 32);
  v += __shfl_xor(v, 8, 32);
  return v;
}

// ---------- kernel 1a: column sums of x (for x_bar) ----------
__global__ void k_colsum(const float* __restrict__ x, float* __restrict__ xbar) {
  int d = threadIdx.x;
  int r0 = blockIdx.x * 32;
  float s = 0.f;
#pragma unroll 4
  for (int i = 0; i < 32; ++i) s += x[(size_t)(r0 + i) * DD + d];
  atomicAdd(&xbar[d], s);
}

// ---------- kernel 1b: gk = x_bar@gK, gv = x_bar@gVw+gVb, wg = gQ@gk ----------
__global__ void k_globals(const float* __restrict__ xbarsum,
                          const float* __restrict__ gK, const float* __restrict__ gVw,
                          const float* __restrict__ gVb, const float* __restrict__ gQ,
                          float* __restrict__ gv, float* __restrict__ wg) {
  __shared__ float sx[DD];
  __shared__ float sgk[DD];
  int t = threadIdx.x;
  sx[t] = xbarsum[t] * (1.0f / (float)NN);
  __syncthreads();
  float a = 0.f, b = 0.f;
  for (int d = 0; d < DD; ++d) {
    float xb = sx[d];
    a += xb * gK[d * DD + t];
    b += xb * gVw[d * DD + t];
  }
  sgk[t] = a;
  gv[t] = b + gVb[t];
  __syncthreads();
  float c = 0.f;
  for (int j = 0; j < DD; ++j) c += gQ[t * DD + j] * sgk[j];
  wg[t] = c;
}

// ---------- kernel 2: projections via WMMA ----------
// mode (blockIdx.y): 0 -> lq (bf16 row-major), 1 -> lk (bf16 row-major),
//                    2 -> lv stored transposed lvT[d][n] (bf16), 3 -> ev (f32)
__global__ void __launch_bounds__(128)
k_proj(const float* __restrict__ x,
       const float* __restrict__ lQ, const float* __restrict__ lK,
       const float* __restrict__ lVw, const float* __restrict__ lVb,
       const float* __restrict__ eVw, const float* __restrict__ eVb,
       unsigned short* __restrict__ lq, unsigned short* __restrict__ lk,
       unsigned short* __restrict__ lvT, float* __restrict__ ev) {
  __shared__ __align__(16) unsigned short s_xa[16][40];    // A tile 16x32 (padded)
  __shared__ __align__(16) unsigned short s_wT[DD][40];    // W chunk transposed: [col][k]
  int mode = blockIdx.y;
  const float* W    = (mode == 0) ? lQ : (mode == 1) ? lK : (mode == 2) ? lVw : eVw;
  const float* bias = (mode == 2) ? lVb : (mode == 3) ? eVb : nullptr;
  int q0 = blockIdx.x * 16;
  int tid = threadIdx.x;
  int wv = tid >> 5, lane = tid & 31, hf = lane >> 4, ln = lane & 15;

  v8f acc[4] = {};
  for (int kc = 0; kc < 8; ++kc) {
    int k0 = kc * 32;
    // stage A: x[q0..q0+15][k0..k0+31] -> bf16
    {
      int e0 = tid * 4, r = e0 >> 5, c = e0 & 31;
      const float4 v = *(const float4*)&x[(size_t)(q0 + r) * DD + k0 + c];
      s_xa[r][c + 0] = f2bfu(v.x); s_xa[r][c + 1] = f2bfu(v.y);
      s_xa[r][c + 2] = f2bfu(v.z); s_xa[r][c + 3] = f2bfu(v.w);
    }
    // stage B transposed: W[k0..k0+31][0..255] -> s_wT[col][k]
    {
      int r = tid >> 2, cb = (tid & 3) * 64;
      for (int i = 0; i < 64; i += 4) {
        const float4 v = *(const float4*)&W[(size_t)(k0 + r) * DD + cb + i];
        s_wT[cb + i + 0][r] = f2bfu(v.x); s_wT[cb + i + 1][r] = f2bfu(v.y);
        s_wT[cb + i + 2][r] = f2bfu(v.z); s_wT[cb + i + 3][r] = f2bfu(v.w);
      }
    }
    __syncthreads();
    v16bf a;
#pragma unroll
    for (int e = 0; e < 16; ++e) a[e] = ubf(s_xa[ln][kmap(e, hf)]);
#pragma unroll
    for (int ct = 0; ct < 4; ++ct) {
      int col0 = wv * 64 + ct * 16;
      v16bf b;
#pragma unroll
      for (int e = 0; e < 16; ++e) b[e] = ubf(s_wT[col0 + ln][kmap(e, hf)]);
      acc[ct] = __builtin_amdgcn_wmma_f32_16x16x32_bf16(false, a, false, b,
                                                        (short)0, acc[ct], false, false);
    }
    __syncthreads();
  }
#pragma unroll
  for (int ct = 0; ct < 4; ++ct) {
    int col = wv * 64 + ct * 16 + ln;
    float bv = bias ? bias[col] : 0.f;
#pragma unroll
    for (int v = 0; v < 8; ++v) {
      int row = hf * 8 + v;
      size_t i = (size_t)(q0 + row);
      float val = acc[ct][v] + bv;
      if (mode == 0)      lq[i * DD + col]  = f2bfu(val);
      else if (mode == 1) lk[i * DD + col]  = f2bfu(val);
      else if (mode == 2) lvT[(size_t)col * NN + i] = f2bfu(val);
      else                ev[i * DD + col]  = val;
    }
  }
}

// ---------- kernel 3: es[i] = x[i].esW ; gs[i] = scale * x[i].wg ----------
__global__ void k_esgs(const float* __restrict__ x, const float* __restrict__ esW,
                       const float* __restrict__ wg,
                       float* __restrict__ es, float* __restrict__ gs) {
  int wv = threadIdx.x >> 5, lane = threadIdx.x & 31;
  int row = blockIdx.x * 8 + wv;
  float a = 0.f, b = 0.f;
#pragma unroll
  for (int c = lane; c < DD; c += 32) {
    float xv = x[(size_t)row * DD + c];
    a += xv * esW[c];
    b += xv * wg[c];
  }
  for (int off = 16; off; off >>= 1) {
    a += __shfl_xor(a, off, 32);
    b += __shfl_xor(b, off, 32);
  }
  if (lane == 0) { es[row] = a; gs[row] = b * 0.0625f; }
}

// ---------- kernel 4: fused masked flash attention with 3-way softmax ----------
// block = 128 threads (4 waves) = 32 query rows. Keys processed 32 at a time.
// wave w: qs = w>>1 selects query sub-tile (16 rows), kh = w&1 selects key half
// (score phase) and output-column half (PV phase).
__global__ void __launch_bounds__(128)
k_flash(const uint8_t* __restrict__ adj,
        const unsigned short* __restrict__ lq, const unsigned short* __restrict__ lk,
        const unsigned short* __restrict__ lvT, const float* __restrict__ ev,
        const float* __restrict__ es, const float* __restrict__ gs,
        const float* __restrict__ gv, float* __restrict__ out) {
  __shared__ __align__(16) unsigned short s_lq[32][264];  // Q tile 32x256
  __shared__ __align__(16) unsigned short s_lk[32][264];  // K tile 32x256
  __shared__ __align__(16) unsigned short s_lv[DD][40];   // V^T tile 256x32
  __shared__ __align__(16) unsigned short s_p[32][40];    // probs 32x32 bf16
  __shared__ float s_m[32], s_l[32], s_r[32];
  __shared__ float s_wmax[4][16], s_wsum[4][16];

  int q0 = blockIdx.x * 32;
  int tid = threadIdx.x;
  int wv = tid >> 5, lane = tid & 31, hf = lane >> 4, ln = lane & 15;
  int qs = wv >> 1, kh = wv & 1;
  int qb = qs * 16;           // query sub-tile base row (block-local)
  int kk0 = kh * 16;          // key sub-tile base within 32-key chunk

  // stage Q tile once (reused for all 256 key chunks)
  {
    int r = tid >> 2, cb = (tid & 3) * 64;
    for (int i = 0; i < 64; i += 8)
      cp16(&s_lq[r][cb + i], &lq[(size_t)(q0 + r) * DD + cb + i]);
  }
  // 3-way softmax init: global + essential scores join the max/denominator
  if (tid < 32) {
    float g = gs[q0 + tid], e = es[q0 + tid];
    float m0 = fmaxf(g, e);
    s_m[tid] = m0;
    s_l[tid] = __expf(g - m0) + __expf(e - m0);
  }

  v8f acc[8] = {};

  for (int k0 = 0; k0 < NN; k0 += 32) {
    // stage K tile: lk[k0..k0+31][0..255]
    {
      int r = tid >> 2, cb = (tid & 3) * 64;
      for (int i = 0; i < 64; i += 8)
        cp16(&s_lk[r][cb + i], &lk[(size_t)(k0 + r) * DD + cb + i]);
    }
    // stage V^T tile: lvT[0..255][k0..k0+31]
#pragma unroll
    for (int j = 0; j < 2; ++j) {
      int d = tid + j * 128;
      cp16(&s_lv[d][0],  &lvT[(size_t)d * NN + k0 + 0]);
      cp16(&s_lv[d][8],  &lvT[(size_t)d * NN + k0 + 8]);
      cp16(&s_lv[d][16], &lvT[(size_t)d * NN + k0 + 16]);
      cp16(&s_lv[d][24], &lvT[(size_t)d * NN + k0 + 24]);
    }
    if (k0 + 32 < NN) {  // pull next tiles toward L2 (global_prefetch_b8)
      __builtin_prefetch(&lk[(size_t)(k0 + 32 + (tid >> 2)) * DD + (tid & 3) * 64], 0, 3);
      __builtin_prefetch(&lvT[(size_t)(tid * 2) * NN + k0 + 32], 0, 3);
    }
    cp_fence();
    __syncthreads();

    // ---- scores: this wave's 16x16 (q-sub x key-sub) tile, K=256, 8 WMMAs ----
    v8f sc = {};
#pragma unroll
    for (int dc = 0; dc < 8; ++dc) {
      v16bf a, b;
#pragma unroll
      for (int e = 0; e < 16; ++e) {
        int k = kmap(e, hf);
        a[e] = ubf(s_lq[qb + ln][dc * 32 + k]);     // A[m=ln][k]
        b[e] = ubf(s_lk[kk0 + ln][dc * 32 + k]);    // B[k][n=ln] = lk[key n][d k]
      }
      sc = __builtin_amdgcn_wmma_f32_16x16x32_bf16(false, a, false, b,
                                                   (short)0, sc, false, false);
    }

    // ---- mask + per-row max ----
    int key = k0 + kk0 + ln;
    float sv[8];
    bool msk[8];
    float tm[8];
#pragma unroll
    for (int v = 0; v < 8; ++v) {
      int row = qb + hf * 8 + v;
      msk[v] = adj[(size_t)(q0 + row) * NN + key] != 0;
      sv[v] = msk[v] ? sc[v] * 0.0625f : -1e30f;    // scale = 1/sqrt(256)
      tm[v] = rowmax16(sv[v]);
    }
    if (ln == 0) {
#pragma unroll
      for (int v = 0; v < 8; ++v) s_wmax[wv][hf * 8 + v] = tm[v];
    }
    __syncthreads();

    if (tid < 32) {
      int t = tid, qq = t >> 4, r = t & 15;
      float nm = fmaxf(s_m[t], fmaxf(s_wmax[2 * qq][r], s_wmax[2 * qq + 1][r]));
      s_r[t] = __expf(s_m[t] - nm);
      s_m[t] = nm;
      s_l[t] *= s_r[t];
    }
    __syncthreads();

    // ---- rescale acc, exponentiate, row sums, write P ----
#pragma unroll
    for (int v = 0; v < 8; ++v) {
      int rl = hf * 8 + v;
      int row = qb + rl;
      float rr = s_r[row];
#pragma unroll
      for (int ct = 0; ct < 8; ++ct) acc[ct][v] *= rr;
      float e = msk[v] ? __expf(sv[v] - s_m[row]) : 0.f;
      s_p[row][kk0 + ln] = f2bfu(e);
      float rs = rowsum16(e);
      if (ln == 0) s_wsum[wv][rl] = rs;
    }
    __syncthreads();
    if (tid < 32) {
      int qq = tid >> 4, r = tid & 15;
      s_l[tid] += s_wsum[2 * qq][r] + s_wsum[2 * qq + 1][r];
    }

    // ---- PV: acc += P(16x32) x V(32 keys x 128 cols per wave) ----
    v16bf pa;
#pragma unroll
    for (int e = 0; e < 16; ++e) pa[e] = ubf(s_p[qb + ln][kmap(e, hf)]);
#pragma unroll
    for (int ct = 0; ct < 8; ++ct) {
      int col0 = kh * 128 + ct * 16;
      v16bf b;
#pragma unroll
      for (int e = 0; e < 16; ++e) b[e] = ubf(s_lv[col0 + ln][kmap(e, hf)]);
      acc[ct] = __builtin_amdgcn_wmma_f32_16x16x32_bf16(false, pa, false, b,
                                                        (short)0, acc[ct], false, false);
    }
    __syncthreads();
  }

  // ---- epilogue: add global + essential value branches, normalize ----
#pragma unroll
  for (int ct = 0; ct < 8; ++ct) {
    int col = kh * 128 + ct * 16 + ln;
    float gvv = gv[col];
#pragma unroll
    for (int v = 0; v < 8; ++v) {
      int row = qb + hf * 8 + v;
      size_t i = (size_t)(q0 + row);
      float ge = __expf(gs[i] - s_m[row]);
      float ee = __expf(es[i] - s_m[row]);
      out[i * DD + col] = (acc[ct][v] + ge * gvv + ee * ev[i * DD + col]) / s_l[row];
    }
  }
}

// ---------- host launcher ----------
extern "C" void kernel_launch(void* const* d_in, const int* in_sizes, int n_in,
                              void* d_out, int out_size, void* d_ws, size_t ws_size,
                              hipStream_t stream) {
  (void)in_sizes; (void)n_in; (void)out_size; (void)ws_size;
  const uint8_t* adj = (const uint8_t*)d_in[0];
  const float* x   = (const float*)d_in[1];
  const float* lQ  = (const float*)d_in[2];
  const float* lK  = (const float*)d_in[3];
  const float* lVw = (const float*)d_in[4];
  const float* lVb = (const float*)d_in[5];
  const float* gQ  = (const float*)d_in[6];
  const float* gK  = (const float*)d_in[7];
  const float* gVw = (const float*)d_in[8];
  const float* gVb = (const float*)d_in[9];
  const float* esW = (const float*)d_in[10];
  const float* eVw = (const float*)d_in[11];
  const float* eVb = (const float*)d_in[12];
  float* out = (float*)d_out;

  char* w = (char*)d_ws;
  auto take = [&](size_t bytes) {
    char* p = w;
    w += (bytes + 255) & ~(size_t)255;
    return p;
  };
  float* xbar = (float*)take(DD * sizeof(float));
  float* gv   = (float*)take(DD * sizeof(float));
  float* wg   = (float*)take(DD * sizeof(float));
  float* es   = (float*)take(NN * sizeof(float));
  float* gs   = (float*)take(NN * sizeof(float));
  unsigned short* lq  = (unsigned short*)take((size_t)NN * DD * 2);
  unsigned short* lk  = (unsigned short*)take((size_t)NN * DD * 2);
  unsigned short* lvT = (unsigned short*)take((size_t)NN * DD * 2);
  float* ev = (float*)take((size_t)NN * DD * sizeof(float));

  (void)hipMemsetAsync(xbar, 0, DD * sizeof(float), stream);
  k_colsum<<<NN / 32, DD, 0, stream>>>(x, xbar);
  k_globals<<<1, DD, 0, stream>>>(xbar, gK, gVw, gVb, gQ, gv, wg);
  k_proj<<<dim3(NN / 16, 4), 128, 0, stream>>>(x, lQ, lK, lVw, lVb, eVw, eVb,
                                               lq, lk, lvT, ev);
  k_esgs<<<NN / 8, 256, 0, stream>>>(x, esW, wg, es, gs);
  k_flash<<<NN / 32, 128, 0, stream>>>(adj, lq, lk, lvT, ev, es, gs, gv, out);
}
